// Attention_layer_55147380081228
// MI455X (gfx1250) — compile-verified, matrix operand order checked by reading, not run on hardware
//
#include <hip/hip_runtime.h>

// ---------------------------------------------------------------------------
// Attention layer for MI455X (gfx1250, wave32, WMMA + TDM).
//   q,k,v = X @ W{q,k,v}           (bf16 WMMA, f32 accum)  -> workspace (bf16)
//   out   = softmax(q k^T / sqrt(D)) @ v                   (flash, bf16 WMMA)
// The reference's mask is added along the QUERY axis (constant per softmax
// row) -> mathematically a no-op under softmax; omitted.
//
// All WMMA fragments load as two aligned 128-bit LDS loads (v16bf halves 0..7
// = K 8h..8h+7, halves 8..15 = K 16+8h..+7, contiguous runs).  K tiles are
// staged by the Tensor Data Mover (tensor_load_to_lds) with hardware LDS row
// padding; V tiles are register-transposed into LDS by the waves while the
// TDM runs, then the issuing wave drains TENSORcnt before the barrier.
// ---------------------------------------------------------------------------

typedef __attribute__((ext_vector_type(16))) __bf16   v16bf;
typedef __attribute__((ext_vector_type(8)))  float    v8f;
typedef __attribute__((ext_vector_type(4)))  uint32_t u32x4;
typedef __attribute__((ext_vector_type(8)))  uint32_t u32x8;

static constexpr int Bn = 2;
static constexpr int Sn = 4096;
static constexpr int Dn = 128;
#define SCALE 0.0883883476483184405f /* 1/sqrt(128) */

static constexpr int KP = 136; // padded row stride (halves): 256B row + 16B pad
static constexpr int VP = 72;  // padded LDS row stride (halves) for [128][64] V^T
static constexpr int WP = 136; // padded LDS row stride (halves) for [128][128] W^T
static constexpr int PP = 40;  // padded LDS row stride (halves) for [16][32] P

union FragB {
  v16bf v;
  uint4 q[2];
};
union U128B {
  uint4  q;
  __bf16 h[8];
};

__device__ __forceinline__ float redmax16(float v) {
  v = fmaxf(v, __shfl_xor(v, 1, 32));
  v = fmaxf(v, __shfl_xor(v, 2, 32));
  v = fmaxf(v, __shfl_xor(v, 4, 32));
  v = fmaxf(v, __shfl_xor(v, 8, 32));
  return v;
}
__device__ __forceinline__ float redsum16(float v) {
  v += __shfl_xor(v, 1, 32);
  v += __shfl_xor(v, 2, 32);
  v += __shfl_xor(v, 4, 32);
  v += __shfl_xor(v, 8, 32);
  return v;
}

// ---------------------------------------------------------------------------
// Kernel 1: QKV projection.  One wave per 16-row tile, 8 waves per block.
// Each weight staged TRANSPOSED (f32 -> bf16) in LDS, reused by all 8 waves.
// ---------------------------------------------------------------------------
__global__ __launch_bounds__(256)
void qkv_proj_kernel(const float* __restrict__ X,
                     const float* __restrict__ Wq,
                     const float* __restrict__ Wk,
                     const float* __restrict__ Wv,
                     __bf16* __restrict__ qo,
                     __bf16* __restrict__ ko,
                     __bf16* __restrict__ vo) {
  __shared__ __bf16 wldsT[Dn * WP]; // W^T[e][d], ~34 KB

  const int lane = threadIdx.x & 31;
  const int wave = threadIdx.x >> 5;
  const int half = lane >> 4;
  const int mn   = lane & 15;

  const int row0 = blockIdx.x * 128 + wave * 16; // flat row in [0, B*S)

  // A fragments of X: 16x128 f32 -> 4 chunks of 16x32 bf16 (lane = row M).
  v16bf xa[4];
  {
    const float* xrow = X + (size_t)(row0 + mn) * Dn;
#pragma unroll
    for (int c = 0; c < 4; ++c) {
#pragma unroll
      for (int g = 0; g < 2; ++g) { // two contiguous 8-element K runs
        const int k0 = c * 32 + g * 16 + 8 * half;
        const float4 f0 = *(const float4*)(xrow + k0);
        const float4 f1 = *(const float4*)(xrow + k0 + 4);
        xa[c][g * 8 + 0] = (__bf16)f0.x;
        xa[c][g * 8 + 1] = (__bf16)f0.y;
        xa[c][g * 8 + 2] = (__bf16)f0.z;
        xa[c][g * 8 + 3] = (__bf16)f0.w;
        xa[c][g * 8 + 4] = (__bf16)f1.x;
        xa[c][g * 8 + 5] = (__bf16)f1.y;
        xa[c][g * 8 + 6] = (__bf16)f1.z;
        xa[c][g * 8 + 7] = (__bf16)f1.w;
      }
    }
  }

#pragma unroll // full unroll -> per-copy kernarg pointers -> global_store_b16
  for (int w = 0; w < 3; ++w) {
    const float* Wsrc = (w == 0) ? Wq : (w == 1) ? Wk : Wv;
    __bf16*      Odst = (w == 0) ? qo : (w == 1) ? ko : vo;

    __syncthreads();
    // stage W (f32, [d][e]) transposed as bf16 W^T[e][d]
    for (int i = threadIdx.x; i < Dn * Dn; i += 256) {
      const int d = i >> 7, e = i & 127;
      wldsT[e * WP + d] = (__bf16)Wsrc[i];
    }
    __syncthreads();

#pragma unroll
    for (int nt = 0; nt < 8; ++nt) {
      v8f acc = {};
#pragma unroll
      for (int c = 0; c < 4; ++c) {
        FragB bf; // B(k,n) = W[k][nt*16+n] = W^T[nt*16+n][k]; lane = column n
        const __bf16* wr = &wldsT[(nt * 16 + mn) * WP + c * 32 + 8 * half];
        bf.q[0] = *(const uint4*)(wr);
        bf.q[1] = *(const uint4*)(wr + 16);
        acc = __builtin_amdgcn_wmma_f32_16x16x32_bf16(
            false, xa[c], false, bf.v, (short)0, acc, false, false);
      }
      // C layout: VGPR r -> row (r + 8*half), col = mn
      __bf16* ocol = Odst + (size_t)row0 * Dn + nt * 16 + mn;
#pragma unroll
      for (int r = 0; r < 8; ++r)
        ocol[(size_t)(r + 8 * half) * Dn] = (__bf16)acc[r];
    }
  }
}

// ---------------------------------------------------------------------------
// Kernel 2: flash attention.  8 waves/block, one 16-query tile per wave.
// K tile staged by TDM (hardware row padding to KP); V tile register-
// transposed into LDS concurrently.  Per 32-key step: 8 WMMA (QK^T) ->
// online softmax -> P via wave-private LDS bounce -> 8 WMMA (PV).
// ---------------------------------------------------------------------------
__global__ __launch_bounds__(256)
void flash_attn_kernel(const __bf16* __restrict__ qb,
                       const __bf16* __restrict__ kb,
                       const __bf16* __restrict__ vb,
                       float* __restrict__ out) {
  __shared__ __bf16 klds[64 * KP];     // K[key][d],  TDM-written, ~17 KB
  __shared__ __bf16 vldsT[Dn * VP];    // V^T[d][key], ~18 KB
  __shared__ __bf16 plds[8 * 16 * PP]; // per-wave P staging, ~10 KB

  const int lane = threadIdx.x & 31;
  const int wave = threadIdx.x >> 5;
  const int half = lane >> 4;
  const int mn   = lane & 15;

  const int b  = blockIdx.x >> 5; // 32 blocks per batch
  const int qc = blockIdx.x & 31;
  const int qrow0 = qc * 128 + wave * 16; // within batch
  const size_t base = (size_t)b * Sn * Dn;

  // Q A-fragments for this wave's 16 rows (4 chunks of 16x32 bf16)
  v16bf qa[4];
  {
    const __bf16* qrow = qb + base + (size_t)(qrow0 + mn) * Dn;
#pragma unroll
    for (int c = 0; c < 4; ++c) {
      FragB f;
      f.q[0] = *(const uint4*)(qrow + c * 32 + 8 * half);
      f.q[1] = *(const uint4*)(qrow + c * 32 + 16 + 8 * half);
      qa[c] = f.v;
    }
  }

  // TDM descriptor pieces that are loop-invariant.
  const uint32_t klds_addr = (uint32_t)(uintptr_t)&klds[0];
  // group1: data_size=1 (2B) | pad_enable | pad_interval=5 (64 DW = 256B row)
  //         | pad_amount=3 (4 DW = 16B)  -> LDS row stride 272B = KP halves
  const uint32_t g1d0 = (1u << 16) | (1u << 20) | (5u << 22) | (3u << 25);

  float m_i[8], l_i[8];
  v8f acc[8];
#pragma unroll
  for (int r = 0; r < 8; ++r) { m_i[r] = -1e30f; l_i[r] = 0.0f; }
#pragma unroll
  for (int nt = 0; nt < 8; ++nt) { v8f z = {}; acc[nt] = z; }

  for (int kt = 0; kt < Sn; kt += 64) {
    __syncthreads(); // previous K/V tile fully consumed

    // ---- K tile via Tensor Data Mover (one wave issues; EXEC ignored) -----
    if (wave == 0) {
      const uint64_t gaddr = (uint64_t)(uintptr_t)(kb + base + (size_t)kt * Dn);
      u32x4 g0;
      g0[0] = 1u;                                     // count=1 (valid D#)
      g0[1] = klds_addr;                              // lds_addr (bytes)
      g0[2] = (uint32_t)gaddr;                        // global_addr[31:0]
      g0[3] = (uint32_t)(gaddr >> 32) | (2u << 30);   // addr[56:32] | type=2
      u32x8 g1;
      g1[0] = g1d0;            // mask=0, data_size=2B, pad cfg
      g1[1] = 128u << 16;      // tensor_dim0 = 128 elements / row
      g1[2] = (4096u & 0xffffu) << 16;  // tensor_dim0 hi=0 | tensor_dim1 lo
      g1[3] = (128u << 16);    // tensor_dim1 hi=0 | tile_dim0 = 128
      g1[4] = 64u;             // tile_dim1 = 64 rows, tile_dim2 = 0
      g1[5] = 128u;            // tensor_dim0_stride = 128 elements
      g1[6] = 0u;              // stride0 hi | stride1 lo
      g1[7] = 0u;              // stride1 hi
      asm volatile("tensor_load_to_lds %0, %1" :: "s"(g0), "s"(g1) : "memory");
    }

    // ---- V tile register-transposed into LDS (all waves, overlaps TDM) ----
    {
      const uint4* vg = (const uint4*)(vb + base + (size_t)kt * Dn);
      for (int i = threadIdx.x; i < 1024; i += 256) { // 64*128 halves / 8
        const int r = i >> 4, c8 = (i & 15) * 8;
        U128B u; u.q = vg[i];
#pragma unroll
        for (int j = 0; j < 8; ++j) vldsT[(c8 + j) * VP + r] = u.h[j];
      }
    }

    if (wave == 0) __builtin_amdgcn_s_wait_tensorcnt(0); // K tile landed
    __syncthreads();

#pragma unroll
    for (int sub = 0; sub < 2; ++sub) {
      const int key0 = sub * 32;

      // ---- scores: two 16x16 fragments over 32 keys -----------------------
      v8f s0 = {}, s1 = {};
#pragma unroll
      for (int c = 0; c < 4; ++c) {
        FragB bk0, bk1; // B(k,n) = K[key n][d k]; lane = key column
        const __bf16* kr0 = &klds[(key0 + mn) * KP + c * 32 + 8 * half];
        const __bf16* kr1 = &klds[(key0 + 16 + mn) * KP + c * 32 + 8 * half];
        bk0.q[0] = *(const uint4*)(kr0);
        bk0.q[1] = *(const uint4*)(kr0 + 16);
        bk1.q[0] = *(const uint4*)(kr1);
        bk1.q[1] = *(const uint4*)(kr1 + 16);
        s0 = __builtin_amdgcn_wmma_f32_16x16x32_bf16(
            false, qa[c], false, bk0.v, (short)0, s0, false, false);
        s1 = __builtin_amdgcn_wmma_f32_16x16x32_bf16(
            false, qa[c], false, bk1.v, (short)0, s1, false, false);
      }

      // ---- online softmax (row m = r + 8*half lives across 16 lanes) ------
      __bf16* pw = &plds[wave * 16 * PP];
#pragma unroll
      for (int r = 0; r < 8; ++r) {
        const float v0 = s0[r] * SCALE;
        const float v1 = s1[r] * SCALE;
        const float mx   = redmax16(fmaxf(v0, v1));
        const float mnew = fmaxf(m_i[r], mx);
        const float corr = __expf(m_i[r] - mnew);
        m_i[r] = mnew;
        const float p0 = __expf(v0 - mnew);
        const float p1 = __expf(v1 - mnew);
        l_i[r] = l_i[r] * corr + redsum16(p0 + p1);
        const int m = r + 8 * half;
        pw[m * PP + mn]      = (__bf16)p0;
        pw[m * PP + mn + 16] = (__bf16)p1;
#pragma unroll
        for (int nt = 0; nt < 8; ++nt) acc[nt][r] *= corr;
      }

      // P staging is wave-private: drain the LDS counter (CDNA5 split wait).
      asm volatile("s_wait_dscnt 0" ::: "memory");

      // ---- P as A fragment (16x32), then acc += P @ V ---------------------
      FragB pa;
      {
        const __bf16* pr = pw + mn * PP + 8 * half;
        pa.q[0] = *(const uint4*)(pr);
        pa.q[1] = *(const uint4*)(pr + 16);
      }
#pragma unroll
      for (int nt = 0; nt < 8; ++nt) {
        FragB bv; // B(k,n) = V[key0+k][nt*16+n] = V^T[nt*16+n][key0+k]
        const __bf16* vr = &vldsT[(nt * 16 + mn) * VP + key0 + 8 * half];
        bv.q[0] = *(const uint4*)(vr);
        bv.q[1] = *(const uint4*)(vr + 16);
        acc[nt] = __builtin_amdgcn_wmma_f32_16x16x32_bf16(
            false, pa.v, false, bv.v, (short)0, acc[nt], false, false);
      }
    }
  }

  // ---- epilogue: out = acc / l ---------------------------------------------
  float inv[8];
#pragma unroll
  for (int r = 0; r < 8; ++r) inv[r] = 1.0f / l_i[r];
  float* obase = out + base + (size_t)qrow0 * Dn;
#pragma unroll
  for (int nt = 0; nt < 8; ++nt)
#pragma unroll
    for (int r = 0; r < 8; ++r) {
      const int m = r + 8 * half;
      obase[(size_t)m * Dn + nt * 16 + mn] = acc[nt][r] * inv[r];
    }
}

// ---------------------------------------------------------------------------
extern "C" void kernel_launch(void* const* d_in, const int* in_sizes, int n_in,
                              void* d_out, int out_size, void* d_ws,
                              size_t ws_size, hipStream_t stream) {
  const float* X  = (const float*)d_in[0]; // [B,S,D] f32
  const float* Wq = (const float*)d_in[4]; // [D,D] f32
  const float* Wk = (const float*)d_in[5];
  const float* Wv = (const float*)d_in[6];
  float* out = (float*)d_out; // [B,S,D] f32

  const size_t n = (size_t)Bn * Sn * Dn;
  __bf16* q = (__bf16*)d_ws; // 3 x 2 MB bf16 staging in workspace
  __bf16* k = q + n;
  __bf16* v = k + n;

  qkv_proj_kernel<<<dim3(Bn * Sn / 128), dim3(256), 0, stream>>>(
      X, Wq, Wk, Wv, q, k, v);
  flash_attn_kernel<<<dim3(Bn * Sn / 128), dim3(256), 0, stream>>>(
      q, k, v, out);
}
